// kv_cache_group_query_78812649882270
// MI455X (gfx1250) — matrix operand order repaired and checked by reading, not compiled, not run on hardware
//
#include <hip/hip_runtime.h>

// ---------------------------------------------------------------------------
// GQA attention with KV cache on MI455X (gfx1250, wave32)
//   - all GEMMs via v_wmma_f32_16x16x32_f16
//   - GEMM tile staging via global_load_async_to_lds_b128 (ASYNCcnt)
//   - attention K-chunk staging via TDM tensor_load_to_lds (TENSORcnt)
// ---------------------------------------------------------------------------

typedef __attribute__((ext_vector_type(16))) _Float16     v16h;
typedef __attribute__((ext_vector_type(8)))  float        v8f;
typedef __attribute__((ext_vector_type(4)))  unsigned int v4u;
typedef __attribute__((ext_vector_type(8)))  int          v8i;
typedef __attribute__((ext_vector_type(4)))  int          v4i;

union Frag16 {           // 16 halves = 8 dwords (A or B operand of 16x16x32)
    v16h h;
    unsigned int u[8];
};

#define NUM_Q   32
#define NUM_KV  8
#define HD      128
#define TT      1024      // T
#define BB      2         // B
#define CC      4096      // C
#define SS      2048      // start_pos + T
#define STARTP  1024

static __device__ __forceinline__ unsigned lds_off32(const void* p) {
    // generic pointer to LDS: low 32 bits = byte offset in the wave's LDS alloc
    return (unsigned)(unsigned long long)p;
}

static __device__ __forceinline__ void async_b128(unsigned lds, const void* g) {
    asm volatile("global_load_async_to_lds_b128 %0, %1, off"
                 :: "v"(lds), "v"((unsigned long long)g) : "memory");
}

// ---------------------------------------------------------------------------
// f32 -> f16 conversion (one pass; weights & activations)
// ---------------------------------------------------------------------------
__global__ void cvt_f16(const float* __restrict__ src,
                        _Float16* __restrict__ dst, unsigned n)
{
    unsigned i = blockIdx.x * 256 + threadIdx.x;
    if (i < n) dst[i] = (_Float16)src[i];
}

// ---------------------------------------------------------------------------
// GEMM:  Y[M,N] = A[M,K] * W[N,K]^T    (A,W: f16, Y: f32)
// 256 threads (8 waves); block tile 128(M) x 64(N); K step 32.
// Tiles staged with async b128 copies directly into LDS.
// ---------------------------------------------------------------------------
__global__ __launch_bounds__(256)
void gemm_h(const _Float16* __restrict__ A, const _Float16* __restrict__ W,
            float* __restrict__ Y, int M, int N, int K)
{
    __shared__ _Float16 As[128 * 40];   // 80B row stride: 16B-aligned, 64-bank
    __shared__ _Float16 Bs[64 * 40];

    const int tid  = threadIdx.x;
    const int wave = tid >> 5;
    const int lane = tid & 31;
    const int bm   = blockIdx.y * 128;
    const int bn   = blockIdx.x * 64;
    const int wm   = (wave >> 1) * 32;
    const int wn   = (wave & 1) * 32;

    v8f acc[2][2] = {};

    for (int k0 = 0; k0 < K; k0 += 32) {
        __syncthreads();
        // A tile 128x32 halves = 512 b128 transfers -> 2 per thread
        #pragma unroll
        for (int it = 0; it < 2; ++it) {
            int e = it * 256 + tid;            // 0..511
            int r = e >> 2;                    // 0..127
            int c = (e & 3) << 3;              // 0,8,16,24 halves
            async_b128(lds_off32(&As[r * 40 + c]),
                       A + (size_t)(bm + r) * K + k0 + c);
        }
        // W tile 64x32 halves = 256 b128 transfers -> 1 per thread
        {
            int r = tid >> 2;
            int c = (tid & 3) << 3;
            async_b128(lds_off32(&Bs[r * 40 + c]),
                       W + (size_t)(bn + r) * K + k0 + c);
        }
        asm volatile("s_wait_asynccnt 0" ::: "memory");
        __syncthreads();

        // fragments per CDNA5 ISA 7.12.2 layouts
        Frag16 a[2], b[2];
        const int khalf = (lane >> 4) << 3;
        #pragma unroll
        for (int mt = 0; mt < 2; ++mt) {
            int row = wm + mt * 16 + (lane & 15);
            #pragma unroll
            for (int v = 0; v < 8; ++v) {
                int kk = ((v >> 2) << 4) + khalf + ((v & 3) << 1);
                a[mt].u[v] = *(const unsigned int*)&As[row * 40 + kk];
            }
        }
        const int kb = (lane >> 4) << 4;
        #pragma unroll
        for (int nt = 0; nt < 2; ++nt) {
            int row = wn + nt * 16 + (lane & 15);
            #pragma unroll
            for (int v = 0; v < 8; ++v)
                b[nt].u[v] = *(const unsigned int*)&Bs[row * 40 + ((v >> 2) << 4) + khalf + ((v & 3) << 1)];
        }
        #pragma unroll
        for (int mt = 0; mt < 2; ++mt)
            #pragma unroll
            for (int nt = 0; nt < 2; ++nt)
                acc[mt][nt] = __builtin_amdgcn_wmma_f32_16x16x32_f16(
                    false, a[mt].h, false, b[nt].h, (short)0, acc[mt][nt],
                    false, false);
    }

    #pragma unroll
    for (int mt = 0; mt < 2; ++mt)
        #pragma unroll
        for (int nt = 0; nt < 2; ++nt)
            #pragma unroll
            for (int r = 0; r < 8; ++r) {
                int row = bm + wm + mt * 16 + ((lane >> 4) << 3) + r;
                int col = bn + wn + nt * 16 + (lane & 15);
                Y[(size_t)row * N + col] = acc[mt][nt][r];
            }
}

// ---------------------------------------------------------------------------
// RoPE: rotate Q pairs -> q_f16, rotate new K pairs -> kfull_f16[s>=1024]
// ---------------------------------------------------------------------------
__global__ void rope_pack(const float* __restrict__ q32,
                          const float* __restrict__ k32,
                          _Float16* __restrict__ qf,
                          _Float16* __restrict__ kfull)
{
    size_t idx = (size_t)blockIdx.x * blockDim.x + threadIdx.x;
    const size_t total = (size_t)BB * TT * (NUM_Q + NUM_KV) * (HD / 2);
    if (idx >= total) return;
    int i = idx & 63;
    int h = (int)((idx >> 6) % (NUM_Q + NUM_KV));
    int t = (int)((idx >> 6) / (NUM_Q + NUM_KV) % TT);
    int b = (int)(idx / ((size_t)64 * (NUM_Q + NUM_KV) * TT));

    float pos  = (float)(STARTP + t);
    float freq = pos * __powf(10000.0f, -(float)i / 64.0f);
    float s, c;
    sincosf(freq, &s, &c);

    if (h < NUM_Q) {
        size_t o = (((size_t)b * TT + t) * NUM_Q + h) * HD + 2 * i;
        float x1 = q32[o], x2 = q32[o + 1];
        qf[o]     = (_Float16)(x1 * c - x2 * s);
        qf[o + 1] = (_Float16)(x1 * s + x2 * c);
    } else {
        int kh = h - NUM_Q;
        size_t si = (((size_t)b * TT + t) * NUM_KV + kh) * HD + 2 * i;
        size_t di = (((size_t)b * SS + STARTP + t) * NUM_KV + kh) * HD + 2 * i;
        float x1 = k32[si], x2 = k32[si + 1];
        kfull[di]     = (_Float16)(x1 * c - x2 * s);
        kfull[di + 1] = (_Float16)(x1 * s + x2 * c);
    }
}

// ---------------------------------------------------------------------------
// Assemble f16 K/V caches: s<1024 from input caches, V tail from new V.
// ---------------------------------------------------------------------------
__global__ void build_kv(const float* __restrict__ ck,
                         const float* __restrict__ cv,
                         const float* __restrict__ v32,
                         _Float16* __restrict__ kfull,
                         _Float16* __restrict__ vfull)
{
    size_t idx = (size_t)blockIdx.x * blockDim.x + threadIdx.x;
    const size_t total = (size_t)BB * SS * NUM_KV * HD;
    if (idx >= total) return;
    int d  = idx & 127;
    int kh = (int)((idx >> 7) & 7);
    int s  = (int)((idx >> 10) & (SS - 1));
    int b  = (int)(idx >> 21);
    if (s < STARTP) {
        size_t si = (((size_t)b * 4096 + s) * NUM_KV + kh) * HD + d;
        kfull[idx] = (_Float16)ck[si];
        vfull[idx] = (_Float16)cv[si];
    } else {
        size_t si = (((size_t)b * TT + (s - STARTP)) * NUM_KV + kh) * HD + d;
        vfull[idx] = (_Float16)v32[si];
    }
}

// ---------------------------------------------------------------------------
// Flash attention: one block per (b, q-head, 32-query tile), 2 waves.
// K chunk fetched by the Tensor Data Mover (2D tile, LDS padding -> 136 stride)
// ---------------------------------------------------------------------------
__global__ __launch_bounds__(64)
void attn_kernel(const _Float16* __restrict__ qf,
                 const _Float16* __restrict__ kf,
                 const _Float16* __restrict__ vf,
                 _Float16* __restrict__ ctx)
{
    __shared__ _Float16 Ks[64 * 136];      // filled by TDM (pad 256B + 16B)
    __shared__ _Float16 Vs[128 * 72];      // d-major (transposed)
    __shared__ _Float16 Ps[2 * 16 * 72];   // per-wave probabilities

    int blk    = blockIdx.x;
    int mtile  = blk & 31;
    int h      = (blk >> 5) & 31;
    int b      = blk >> 10;
    int kh     = h >> 2;                   // G = 4
    int m_base = mtile * 32;

    int tid  = threadIdx.x;
    int wave = tid >> 5;
    int lane = tid & 31;
    int wm   = m_base + wave * 16;

    // Q fragments resident in VGPRs for the whole block
    Frag16 qa[4];
    {
        int row = wm + (lane & 15);
        const _Float16* qrow = qf + (((size_t)b * TT + row) * NUM_Q + h) * HD;
        int khalf = (lane >> 4) << 3;
        #pragma unroll
        for (int kc = 0; kc < 4; ++kc)
            #pragma unroll
            for (int v = 0; v < 8; ++v) {
                int d = kc * 32 + ((v >> 2) << 4) + khalf + ((v & 3) << 1);
                qa[kc].u[v] = *(const unsigned int*)&qrow[d];
            }
    }

    float mrun[8], lrun[8];
    v8f O[8] = {};
    #pragma unroll
    for (int r = 0; r < 8; ++r) { mrun[r] = -3.0e38f; lrun[r] = 0.0f; }

    const int s_limit = STARTP + m_base + 32;
    const int nchunks = (s_limit + 63) >> 6;
    const float scale = 0.08838834764831845f;   // 1/sqrt(128)
    const unsigned ks_lds = lds_off32(&Ks[0]);

    for (int ch = 0; ch < nchunks; ++ch) {
        int s0 = ch * 64;
        __syncthreads();

        // ---- TDM: K chunk = 2D tile 128(d) x 64(s), row stride 1024 elems,
        //      LDS pad 16B every 256B -> row stride 136 halves. wave0 issues.
        if (tid < 32) {
            unsigned long long ga = (unsigned long long)
                (kf + (((size_t)b * SS + s0) * NUM_KV + kh) * HD);
            v4u g0;
            g0.x = 1u;                                   // count=1, load desc
            g0.y = ks_lds;                               // lds_addr
            g0.z = (unsigned)ga;                         // global_addr lo
            g0.w = (unsigned)((ga >> 32) & 0x01FFFFFFull) | (2u << 30); // type=2
            v8i g1;
            g1[0] = 0x07510000;   // data_size=2B | pad_en | interval=256B | amt=16B
            g1[1] = (int)(128u << 16);   // tensor_dim0 = 128   (bits 48..63)
            g1[2] = (int)(2048u << 16);  // tensor_dim1 = 2048  (bits 80..95)
            g1[3] = (int)(128u << 16);   // tile_dim0 = 128     (bits 112..127)
            g1[4] = 64;                  // tile_dim1 = 64, tile_dim2 = 0
            g1[5] = 1024;                // tensor_dim0_stride = 8*128 elems
            g1[6] = 0;
            g1[7] = 0;
            v4i g2 = {0, 0, 0, 0}, g3 = {0, 0, 0, 0};
            v8i g4 = {0, 0, 0, 0, 0, 0, 0, 0};           // 6-arg toolchain extra group
            __builtin_amdgcn_tensor_load_to_lds(g0, g1, g2, g3, g4, 0);
        }

        // ---- V chunk load, transposed into d-major LDS (all 64 threads) ----
        #pragma unroll
        for (int it = 0; it < 16; ++it) {
            int e = it * 64 + tid;
            int r = e >> 4;                        // s row 0..63
            int c = (e & 15) << 3;                 // d base
            const _Float16* src = vf +
                (((size_t)b * SS + s0 + r) * NUM_KV + kh) * HD + c;
            #pragma unroll
            for (int j = 0; j < 8; ++j)
                Vs[(c + j) * 72 + r] = src[j];
        }
        if (tid < 32) __builtin_amdgcn_s_wait_tensorcnt(0);
        __syncthreads();

        // ---- scores: Q @ K^T, 4 n-tiles x 4 d-chunks of WMMA ----
        v8f sc[4];
        #pragma unroll
        for (int nt = 0; nt < 4; ++nt) {
            v8f a = {};
            int krow = nt * 16 + (lane & 15);
            int kb   = (lane >> 4) << 4;
            #pragma unroll
            for (int kc = 0; kc < 4; ++kc) {
                Frag16 bf;
                #pragma unroll
                for (int v = 0; v < 8; ++v)
                    bf.u[v] = *(const unsigned int*)
                        &Ks[krow * 136 + kc * 32 + ((v >> 2) << 4) + ((lane >> 4) << 3) + ((v & 3) << 1)];
                a = __builtin_amdgcn_wmma_f32_16x16x32_f16(
                        false, qa[kc].h, false, bf.h, (short)0, a, false, false);
            }
            sc[nt] = a;
        }

        // ---- scale + causal mask (j <= 1024 + i) ----
        #pragma unroll
        for (int nt = 0; nt < 4; ++nt) {
            int s = s0 + nt * 16 + (lane & 15);
            #pragma unroll
            for (int r = 0; r < 8; ++r) {
                int t = wm + ((lane >> 4) << 3) + r;
                float v = sc[nt][r] * scale;
                if (s > STARTP + t) v = -3.0e38f;
                sc[nt][r] = v;
            }
        }

        // ---- online softmax ----
        float mnew[8], alpha[8];
        #pragma unroll
        for (int r = 0; r < 8; ++r) {
            float v = fmaxf(fmaxf(sc[0][r], sc[1][r]),
                            fmaxf(sc[2][r], sc[3][r]));
            #pragma unroll
            for (int off = 1; off < 16; off <<= 1)
                v = fmaxf(v, __shfl_xor(v, off, 32));
            mnew[r]  = fmaxf(mrun[r], v);
            alpha[r] = __expf(mrun[r] - mnew[r]);
            mrun[r]  = mnew[r];
        }
        #pragma unroll
        for (int nt = 0; nt < 4; ++nt)
            #pragma unroll
            for (int r = 0; r < 8; ++r)
                sc[nt][r] = __expf(sc[nt][r] - mnew[r]);
        #pragma unroll
        for (int r = 0; r < 8; ++r) {
            float sum = sc[0][r] + sc[1][r] + sc[2][r] + sc[3][r];
            #pragma unroll
            for (int off = 1; off < 16; off <<= 1)
                sum += __shfl_xor(sum, off, 32);
            lrun[r] = lrun[r] * alpha[r] + sum;
        }
        #pragma unroll
        for (int nt = 0; nt < 8; ++nt)
            #pragma unroll
            for (int r = 0; r < 8; ++r)
                O[nt][r] *= alpha[r];

        // ---- stage P (f16) into this wave's private LDS slab ----
        #pragma unroll
        for (int nt = 0; nt < 4; ++nt)
            #pragma unroll
            for (int r = 0; r < 8; ++r) {
                int rl = ((lane >> 4) << 3) + r;
                int cl = nt * 16 + (lane & 15);
                Ps[(wave * 16 + rl) * 72 + cl] = (_Float16)sc[nt][r];
            }
        asm volatile("s_wait_dscnt 0" ::: "memory");

        // ---- O += P @ V ----
        #pragma unroll
        for (int kc = 0; kc < 2; ++kc) {
            Frag16 pa;
            {
                int row   = wave * 16 + (lane & 15);
                int khalf = (lane >> 4) << 3;
                #pragma unroll
                for (int v = 0; v < 8; ++v) {
                    int ss = kc * 32 + ((v >> 2) << 4) + khalf + ((v & 3) << 1);
                    pa.u[v] = *(const unsigned int*)&Ps[row * 72 + ss];
                }
            }
            int sb = kc * 32 + ((lane >> 4) << 3);
            #pragma unroll
            for (int nt = 0; nt < 8; ++nt) {
                Frag16 bf;
                int d = nt * 16 + (lane & 15);
                #pragma unroll
                for (int v = 0; v < 8; ++v)
                    bf.u[v] = *(const unsigned int*)&Vs[d * 72 + sb + ((v >> 2) << 4) + ((v & 3) << 1)];
                O[nt] = __builtin_amdgcn_wmma_f32_16x16x32_f16(
                            false, pa.h, false, bf.h, (short)0, O[nt],
                            false, false);
            }
        }
    }

    // ---- normalize and write ctx (f16, (b,t,h*128+d)) ----
    #pragma unroll
    for (int r = 0; r < 8; ++r) {
        float inv = 1.0f / lrun[r];
        int t = wm + ((lane >> 4) << 3) + r;
        #pragma unroll
        for (int nt = 0; nt < 8; ++nt) {
            int d = nt * 16 + (lane & 15);
            ctx[((size_t)b * TT + t) * CC + h * HD + d] =
                (_Float16)(O[nt][r] * inv);
        }
    }
}

// ---------------------------------------------------------------------------
// Launch
// ---------------------------------------------------------------------------
extern "C" void kernel_launch(void* const* d_in, const int* in_sizes, int n_in,
                              void* d_out, int out_size, void* d_ws, size_t ws_size,
                              hipStream_t stream)
{
    const float* x       = (const float*)d_in[0];
    // d_in[1] = start_pos (==1024, static)
    const float* cache_k = (const float*)d_in[2];
    const float* cache_v = (const float*)d_in[3];
    const float* Wq      = (const float*)d_in[4];
    const float* Wk      = (const float*)d_in[5];
    const float* Wv      = (const float*)d_in[6];
    const float* Wo      = (const float*)d_in[7];
    float* out = (float*)d_out;

    char* ws = (char*)d_ws;
    _Float16* xh   = (_Float16*)(ws);                         // 16 MB
    _Float16* Wqh  = (_Float16*)(ws + ((size_t)16  << 20));   // 32 MB
    _Float16* Wkh  = (_Float16*)(ws + ((size_t)48  << 20));   //  8 MB
    _Float16* Wvh  = (_Float16*)(ws + ((size_t)56  << 20));   //  8 MB
    _Float16* Woh  = (_Float16*)(ws + ((size_t)64  << 20));   // 32 MB
    float*    q32  = (float*)   (ws + ((size_t)96  << 20));   // 32 MB
    float*    k32  = (float*)   (ws + ((size_t)128 << 20));   //  8 MB
    float*    v32  = (float*)   (ws + ((size_t)136 << 20));   //  8 MB
    _Float16* qfh  = (_Float16*)(ws + ((size_t)144 << 20));   // 16 MB
    _Float16* kful = (_Float16*)(ws + ((size_t)160 << 20));   //  8 MB
    _Float16* vful = (_Float16*)(ws + ((size_t)168 << 20));   //  8 MB
    _Float16* ctx  = (_Float16*)(ws + ((size_t)176 << 20));   // 16 MB

    const int M = BB * TT;   // 2048
    dim3 blk256(256);

    // one-time f32 -> f16 packing of activations + weights
    auto cvt = [&](const float* s, _Float16* d, unsigned n) {
        cvt_f16<<<(n + 255) / 256, 256, 0, stream>>>(s, d, n);
    };
    cvt(x,  xh,  (unsigned)M * CC);
    cvt(Wq, Wqh, (unsigned)CC * CC);
    cvt(Wk, Wkh, (unsigned)(NUM_KV * HD) * CC);
    cvt(Wv, Wvh, (unsigned)(NUM_KV * HD) * CC);
    cvt(Wo, Woh, (unsigned)CC * CC);

    // QKV projections (y = x @ W^T)
    gemm_h<<<dim3(CC / 64, M / 128), blk256, 0, stream>>>(xh, Wqh, q32, M, CC, CC);
    gemm_h<<<dim3((NUM_KV * HD) / 64, M / 128), blk256, 0, stream>>>(xh, Wkh, k32, M, NUM_KV * HD, CC);
    gemm_h<<<dim3((NUM_KV * HD) / 64, M / 128), blk256, 0, stream>>>(xh, Wvh, v32, M, NUM_KV * HD, CC);

    // RoPE + f16 packing
    size_t nrope = (size_t)BB * TT * (NUM_Q + NUM_KV) * (HD / 2);
    rope_pack<<<(unsigned)((nrope + 255) / 256), 256, 0, stream>>>(q32, k32, qfh, kful);

    // f16 KV cache assembly
    size_t nkv = (size_t)BB * SS * NUM_KV * HD;
    build_kv<<<(unsigned)((nkv + 255) / 256), 256, 0, stream>>>(cache_k, cache_v, v32, kful, vful);

    // flash attention
    attn_kernel<<<BB * NUM_Q * (TT / 32), 64, 0, stream>>>(qfh, kful, vful, ctx);

    // output projection
    gemm_h<<<dim3(CC / 64, M / 128), blk256, 0, stream>>>(ctx, Woh, out, M, CC, CC);
}
